// MWT_CZ1d_26963804684331
// MI455X (gfx1250) — matrix operand order repaired and checked by reading, not compiled, not run on hardware
//
#include <hip/hip_runtime.h>
#include <hip/hip_bf16.h>
#include <math.h>

// ---------------- CDNA5 WMMA types ----------------
typedef __attribute__((ext_vector_type(16))) __bf16 v16bf;
typedef __attribute__((ext_vector_type(8)))  __bf16 v8bf;
typedef __attribute__((ext_vector_type(8)))  float  v8f;

#define CK   512          // c*k channel dim
#define NBAT 16           // batch
#define NSEQ 4096         // sequence length
#define NLEV 12           // wavelet levels
#define LMAX 64           // retained modes

// ---------------- workspace layout (bytes) ----------------
#define OFF_XA     ((size_t)0)                       // 128 MiB  X chain ping  [b][i][n] f32
#define OFF_XB     ((size_t)134217728)               //  64 MiB  X chain pong
#define OFF_D      ((size_t)201326592)               //  64 MiB  detail signal / recon ping
#define OFF_XFD_RE ((size_t)268435456)               //  12 MiB  xf(d) re [m][j][b][i] bf16
#define OFF_XFD_IM (OFF_XFD_RE + (size_t)12582912)
#define OFF_XFX_RE (OFF_XFD_IM + (size_t)12582912)
#define OFF_XFX_IM (OFF_XFX_RE + (size_t)12582912)
#define OFF_UDFT   (OFF_XFX_IM + (size_t)12582912)   // 24 MiB [j][b][o][64] u32 = (bf16 re, bf16 im)
#define OFF_USFT   (OFF_UDFT + (size_t)25165824)
#define OFF_UD     (OFF_USFT + (size_t)25165824)     // 128 MiB Ud, all levels, [b][n][i] f32
#define OFF_US     (OFF_UD + (size_t)134184960)      // 128 MiB Us
#define OFF_WB     (OFF_US + (size_t)134184960)      // 6 x 32 MiB bf16 weights [m][o][i]
#define WBSZ       ((size_t)33554432)

static inline __host__ __device__ int lmodes_hd(int j) {
  int h = NSEQ >> (j + 1);
  int l = h / 2 + 1;
  return l > LMAX ? LMAX : l;
}

// ---------------- fragment helpers ----------------
union BF16x16 { v16bf v; v8bf h[2]; __bf16 e[16]; unsigned int u[8]; };
union F32x16  { float4 q[4]; float f[16]; };

// A fragment (16x32 bf16): lane L<16 holds row M=L, K = k0+0..15;
// lane L>=16 holds row M=L-16, K = k0+16..31.
static __device__ __forceinline__ v16bf load_afrag(const __bf16* base, int stride, int k0) {
  int lane = threadIdx.x & 31;
  const __bf16* p = base + (size_t)(lane & 15) * stride + k0 + ((lane & 16) ? 16 : 0);
  BF16x16 f;
  f.h[0] = *(const v8bf*)(p);
  f.h[1] = *(const v8bf*)(p + 8);
  return f.v;
}

// B fragment (32x16 bf16) from bf16 weights in [o][i] (row-major per mode):
// lane L<16: column o0+L, K = k0+0..15 ; lane L>=16: column o0+L-16, K = k0+16..31.
static __device__ __forceinline__ v16bf load_bfrag(const __bf16* base, int o0, int k0) {
  int lane = threadIdx.x & 31;
  const __bf16* p = base + (size_t)(o0 + (lane & 15)) * CK + k0 + ((lane & 16) ? 16 : 0);
  BF16x16 f;
  f.h[0] = *(const v8bf*)(p);
  f.h[1] = *(const v8bf*)(p + 8);
  return f.v;
}

static __device__ __forceinline__ v16bf neg_bf16(v16bf a) {
  BF16x16 f; f.v = a;
#pragma unroll
  for (int t = 0; t < 8; ++t) f.u[t] ^= 0x80008000u;
  return f.v;
}

static __device__ __forceinline__ v8f wmma_bf16(v16bf a, v16bf b, v8f c) {
  return __builtin_amdgcn_wmma_f32_16x16x32_bf16(false, a, false, b, (short)0, c, false, false);
}

static __device__ __forceinline__ unsigned int pack_bf(float re, float im) {
  union { __bf16 b; unsigned short s; } a, c;
  a.b = (__bf16)re; c.b = (__bf16)im;
  return (unsigned int)a.s | ((unsigned int)c.s << 16);
}

// ---------------- kernels ----------------

// x[b][n][i] -> X[b][i][n]  (LDS-tiled transpose)
__global__ void k_transpose_in(const float* __restrict__ in, float* __restrict__ out) {
  __shared__ float tile[32][33];
  int i0 = blockIdx.x * 32, n0 = blockIdx.y * 32, b = blockIdx.z;
#pragma unroll
  for (int rr = 0; rr < 4; ++rr) {
    int n = n0 + threadIdx.y + rr * 8, i = i0 + threadIdx.x;
    tile[threadIdx.y + rr * 8][threadIdx.x] = in[((size_t)b * NSEQ + n) * CK + i];
  }
  __syncthreads();
#pragma unroll
  for (int rr = 0; rr < 4; ++rr) {
    int i = i0 + threadIdx.y + rr * 8, n = n0 + threadIdx.x;
    out[((size_t)b * CK + i) * NSEQ + n] = tile[threadIdx.x][threadIdx.y + rr * 8];
  }
}

// spectral weights: w[i][o][64] f32  ->  wb[m][o][i] bf16 (LDS-tiled i<->m transpose).
// Kills the 256B-stride gather in the GEMM: B fragments become contiguous 16B loads.
__global__ void k_wconv(const float* __restrict__ in, __bf16* __restrict__ out) {
  __shared__ float tile[32][33];
  int i0 = blockIdx.x * 32, m0 = blockIdx.y * 32, o = blockIdx.z;
#pragma unroll
  for (int rr = 0; rr < 4; ++rr) {
    int i = i0 + threadIdx.y + rr * 8;
    tile[threadIdx.y + rr * 8][threadIdx.x] = in[((size_t)i * CK + o) * LMAX + m0 + threadIdx.x];
  }
  __syncthreads();
#pragma unroll
  for (int rr = 0; rr < 4; ++rr) {
    int m = m0 + threadIdx.y + rr * 8;
    out[((size_t)m * CK + o) * CK + i0 + threadIdx.x] =
        (__bf16)tile[threadIdx.x][threadIdx.y + rr * 8];
  }
}

// wavelet analysis: X[b][i][2h] -> d[b][i][h], xs[b][i][h] via 16x8 filters
__global__ void k_analysis(const float* __restrict__ X, float* __restrict__ d,
                           float* __restrict__ xs, const float* __restrict__ ecd,
                           const float* __restrict__ ecs, int halfN) {
  __shared__ float fd[128], fs[128];
  int tid = threadIdx.x;
  if (tid < 128) { fd[tid] = ecd[tid]; fs[tid] = ecs[tid]; }
  __syncthreads();
  int idx = blockIdx.x * 256 + tid;
  int total = NBAT * 64 * halfN;
  if (idx >= total) return;
  int p = idx % halfN, c = (idx / halfN) & 63, b = idx / (halfN * 64);
  int strideIn = halfN * 2;
  float xa[16];
#pragma unroll
  for (int q = 0; q < 16; ++q)
    xa[q] = X[((size_t)b * CK + c * 8 + (q & 7)) * strideIn + 2 * p + (q >> 3)];
#pragma unroll
  for (int kk = 0; kk < 8; ++kk) {
    float sd = 0.f, ss = 0.f;
#pragma unroll
    for (int q = 0; q < 16; ++q) { sd += xa[q] * fd[q * 8 + kk]; ss += xa[q] * fs[q * 8 + kk]; }
    d [((size_t)b * CK + c * 8 + kk) * halfN + p] = sd;
    xs[((size_t)b * CK + c * 8 + kk) * halfN + p] = ss;
  }
}

// truncated forward DFT as WMMA GEMM:
//   xf[m][i] = sum_n sig[b][i][n] * e^{-2pi i m n / Nn},  basis in registers.
// A rows = modes (basis), B cols = channels (signal). One wave per block.
__global__ __launch_bounds__(32)
void k_dft(const float* __restrict__ sig, __bf16* __restrict__ xre,
           __bf16* __restrict__ xim, int Nn, int j, int l) {
  int m0 = blockIdx.x * 16;
  if (m0 >= l) return;
  int itile = blockIdx.y, b = blockIdx.z, lane = threadIdx.x & 31;
  int m = m0 + (lane & 15);
  float w0 = 6.28318530717958647f * (float)m / (float)Nn;
  const float* srow = sig + ((size_t)b * CK + itile * 16 + (lane & 15)) * Nn;
  int kbB = (lane & 16) ? 16 : 0;
  int kLA = (lane & 16) >> 1;       // +8 for upper half lanes (A layout)
  v8f accRe = {}, accIm = {};
  if (Nn >= 32) {
    // fast path: Nn is a multiple of 32 -> unguarded float4 vector loads
    int ksteps = Nn >> 5;
    for (int ks = 0; ks < ksteps; ++ks) {
      int k0 = ks * 32;
      F32x16 sv;
      const float4* sp = (const float4*)(srow + k0 + kbB);
#pragma unroll
      for (int q = 0; q < 4; ++q) sv.q[q] = sp[q];
      BF16x16 bf;
#pragma unroll
      for (int e = 0; e < 16; ++e) bf.e[e] = (__bf16)sv.f[e];
      BF16x16 are, aim;
#pragma unroll
      for (int e = 0; e < 16; ++e) {
        int n = k0 + e + 2 * kLA;
        float s, c;
        __sincosf(w0 * (float)n, &s, &c);
        are.e[e] = (__bf16)c;
        aim.e[e] = (__bf16)(-s);
      }
      accRe = wmma_bf16(are.v, bf.v, accRe);
      accIm = wmma_bf16(aim.v, bf.v, accIm);
    }
  } else {
    // tail path (Nn in {1,2,4,8,16}): clamp index, select value (no exec-masked loads)
    BF16x16 bf;
#pragma unroll
    for (int e = 0; e < 16; ++e) {
      int kk = kbB + e;
      float v = srow[kk < Nn ? kk : 0];
      bf.e[e] = (__bf16)(kk < Nn ? v : 0.f);
    }
    BF16x16 are, aim;
#pragma unroll
    for (int e = 0; e < 16; ++e) {
      int n = e + 2 * kLA;
      float s, c;
      __sincosf(w0 * (float)n, &s, &c);
      are.e[e] = (__bf16)c;
      aim.e[e] = (__bf16)(-s);
    }
    accRe = wmma_bf16(are.v, bf.v, accRe);
    accIm = wmma_bf16(aim.v, bf.v, accIm);
  }
#pragma unroll
  for (int r = 0; r < 8; ++r) {
    int ms = m0 + r + ((lane & 16) ? 8 : 0);
    int is = itile * 16 + (lane & 15);
    size_t o = (((size_t)ms * NLEV + j) * NBAT + b) * CK + is;
    xre[o] = (__bf16)accRe[r];
    xim[o] = (__bf16)accIm[r];
  }
}

__global__ void k_fill0(unsigned int* p, size_t n) {
  size_t i = (size_t)blockIdx.x * 256 + threadIdx.x;
  if (i < n) p[i] = 0u;
}

// batched spectral mix: for each mode m, per level j (batched inside block, weights
// loaded once), complex GEMMs:
//   Ud_ft = d_f @ Aw + x_f @ Bw ;  Us_ft = d_f @ Cw      (M=16 batches, K=512, N=16 tile)
__global__ __launch_bounds__(32)
void k_spectral(const __bf16* __restrict__ xdr, const __bf16* __restrict__ xdi,
                const __bf16* __restrict__ xxr, const __bf16* __restrict__ xxi,
                const __bf16* __restrict__ wAr_, const __bf16* __restrict__ wAi_,
                const __bf16* __restrict__ wBr_, const __bf16* __restrict__ wBi_,
                const __bf16* __restrict__ wCr_, const __bf16* __restrict__ wCi_,
                unsigned int* __restrict__ udft, unsigned int* __restrict__ usft) {
  int m = blockIdx.x, o0 = blockIdx.y * 16, jbase = blockIdx.z * 6;
  int lane = threadIdx.x & 31;
  bool act[6]; bool any = false;
#pragma unroll
  for (int jl = 0; jl < 6; ++jl) { act[jl] = (m < lmodes_hd(jbase + jl)); any |= act[jl]; }
  if (!any) return;
  const __bf16* bAr = wAr_ + (size_t)m * CK * CK;
  const __bf16* bAi = wAi_ + (size_t)m * CK * CK;
  const __bf16* bBr = wBr_ + (size_t)m * CK * CK;
  const __bf16* bBi = wBi_ + (size_t)m * CK * CK;
  const __bf16* bCr = wCr_ + (size_t)m * CK * CK;
  const __bf16* bCi = wCi_ + (size_t)m * CK * CK;
  v8f aUR[6] = {}, aUI[6] = {}, aSR[6] = {}, aSI[6] = {};
#pragma unroll 1
  for (int k0 = 0; k0 < CK; k0 += 32) {
    if (k0 + 32 < CK) {   // CDNA5 speculative prefetch of the next weight block
      size_t pn = (size_t)(o0 + (lane & 15)) * CK + k0 + 32 + ((lane & 16) ? 16 : 0);
      __builtin_prefetch(bAr + pn, 0, 0);  __builtin_prefetch(bAi + pn, 0, 0);
      __builtin_prefetch(bBr + pn, 0, 0);  __builtin_prefetch(bBi + pn, 0, 0);
      __builtin_prefetch(bCr + pn, 0, 0);  __builtin_prefetch(bCi + pn, 0, 0);
    }
    v16bf wAr = load_bfrag(bAr, o0, k0), wAi = load_bfrag(bAi, o0, k0);
    v16bf wBr = load_bfrag(bBr, o0, k0), wBi = load_bfrag(bBi, o0, k0);
    v16bf wCr = load_bfrag(bCr, o0, k0), wCi = load_bfrag(bCi, o0, k0);
#pragma unroll
    for (int jl = 0; jl < 6; ++jl) {
      if (!act[jl]) continue;
      int j = jbase + jl;
      size_t base = ((size_t)m * NLEV + j) * NBAT * CK;   // [m][j][b=row][i]
      v16bf dr = load_afrag(xdr + base, CK, k0);
      v16bf di = load_afrag(xdi + base, CK, k0);
      v16bf xr = load_afrag(xxr + base, CK, k0);
      v16bf xi = load_afrag(xxi + base, CK, k0);
      v16bf ndi = neg_bf16(di), nxi = neg_bf16(xi);
      aUR[jl] = wmma_bf16(dr,  wAr, aUR[jl]);
      aUR[jl] = wmma_bf16(ndi, wAi, aUR[jl]);
      aUR[jl] = wmma_bf16(xr,  wBr, aUR[jl]);
      aUR[jl] = wmma_bf16(nxi, wBi, aUR[jl]);
      aUI[jl] = wmma_bf16(dr,  wAi, aUI[jl]);
      aUI[jl] = wmma_bf16(di,  wAr, aUI[jl]);
      aUI[jl] = wmma_bf16(xr,  wBi, aUI[jl]);
      aUI[jl] = wmma_bf16(xi,  wBr, aUI[jl]);
      aSR[jl] = wmma_bf16(dr,  wCr, aSR[jl]);
      aSR[jl] = wmma_bf16(ndi, wCi, aSR[jl]);
      aSI[jl] = wmma_bf16(dr,  wCi, aSI[jl]);
      aSI[jl] = wmma_bf16(di,  wCr, aSI[jl]);
    }
  }
#pragma unroll
  for (int jl = 0; jl < 6; ++jl) {
    if (!act[jl]) continue;
    int j = jbase + jl;
#pragma unroll
    for (int r = 0; r < 8; ++r) {
      int brow = r + ((lane & 16) ? 8 : 0);
      int oc = o0 + (lane & 15);
      size_t idx = (((size_t)(j * NBAT + brow)) * CK + oc) * LMAX + m;
      udft[idx] = pack_bf(aUR[jl][r], aUI[jl][r]);
      usft[idx] = pack_bf(aSR[jl][r], aSI[jl][r]);
    }
  }
}

// truncated inverse DFT as WMMA GEMM:
//   y[b][n][o] = sum_m w_m * (re cos - im sin), synthesis basis in registers.
__global__ __launch_bounds__(32)
void k_idft(const unsigned int* __restrict__ udft, const unsigned int* __restrict__ usft,
            float* __restrict__ outUd, float* __restrict__ outUs, int Nn, int j, int l) {
  int ntile = blockIdx.x, otile = blockIdx.y;
  int b = blockIdx.z >> 1, sel = blockIdx.z & 1;
  const __bf16* ft = (const __bf16*)(sel ? usft : udft);
  float* out = sel ? outUs : outUd;
  int lane = threadIdx.x & 31;
  const __bf16* abase = ft + ((size_t)((j * NBAT + b) * CK + otile * 16)) * 128;
  int n = ntile * 16 + (lane & 15);
  float wn = 6.28318530717958647f * (float)n / (float)Nn;
  float invN = 1.0f / (float)Nn;
  v8f acc = {};
#pragma unroll 1
  for (int k0 = 0; k0 < 128; k0 += 32) {
    if (k0 >= 2 * l) break;
    v16bf a = load_afrag(abase, 128, k0);     // rows = o, K = interleaved (re,im)
    BF16x16 bf;                               // B: synthesis basis
#pragma unroll
    for (int e = 0; e < 16; ++e) {
      int kk = k0 + ((lane & 16) ? 16 : 0) + e;
      int mm = kk >> 1;
      float wt = (mm >= l || 2 * mm > Nn) ? 0.f
               : ((mm == 0 || 2 * mm == Nn) ? invN : 2.f * invN);
      float s, c;
      __sincosf(wn * (float)mm, &s, &c);
      bf.e[e] = (__bf16)((kk & 1) ? (-wt * s) : (wt * c));
    }
    acc = wmma_bf16(a, bf.v, acc);
  }
#pragma unroll
  for (int r = 0; r < 8; ++r) {
    int os = otile * 16 + r + ((lane & 16) ? 8 : 0);
    int ns = ntile * 16 + (lane & 15);
    if (ns < Nn) out[((size_t)b * Nn + ns) * CK + os] = acc[r];
  }
}

// coarsest-level linear: y = x @ T0w^T + T0b ; writes recon seed [b][0][i]
__global__ void k_t0(const float* __restrict__ X12, const float* __restrict__ T0w,
                     const float* __restrict__ T0b, float* __restrict__ R) {
  int idx = blockIdx.x * 256 + threadIdx.x;
  if (idx >= NBAT * CK) return;
  int b = idx / CK, i = idx % CK, kk = i & 7, cb = i & ~7;
  float acc = T0b[kk];
#pragma unroll
  for (int k = 0; k < 8; ++k) acc += X12[(size_t)b * CK + cb + k] * T0w[kk * 8 + k];
  R[(size_t)b * CK + i] = acc;
}

// reconstruction: x += Us ; cat(x, Ud) @ rc_e / rc_o ; interleave even/odd
__global__ void k_recon(const float* __restrict__ xin, const float* __restrict__ Ud,
                        const float* __restrict__ Us, const float* __restrict__ rce,
                        const float* __restrict__ rco, float* __restrict__ xout, int len) {
  __shared__ float fe[128], fo[128];
  int tid = threadIdx.x;
  if (tid < 128) { fe[tid] = rce[tid]; fo[tid] = rco[tid]; }
  __syncthreads();
  int idx = blockIdx.x * 256 + tid;
  int total = NBAT * 64 * len;
  if (idx >= total) return;
  int p = idx % len, c = (idx / len) & 63, b = idx / (len * 64);
  size_t base = ((size_t)b * len + p) * CK + c * 8;
  float cat[16];
#pragma unroll
  for (int k = 0; k < 8; ++k) cat[k] = xin[base + k] + Us[base + k];
#pragma unroll
  for (int k = 0; k < 8; ++k) cat[8 + k] = Ud[base + k];
  size_t ob = ((size_t)b * (2 * len) + 2 * p) * CK + c * 8;
#pragma unroll
  for (int kk = 0; kk < 8; ++kk) {
    float e = 0.f, o = 0.f;
#pragma unroll
    for (int q = 0; q < 16; ++q) { e += cat[q] * fe[q * 8 + kk]; o += cat[q] * fo[q * 8 + kk]; }
    xout[ob + kk]      = e;
    xout[ob + CK + kk] = o;
  }
}

// ---------------- host orchestration ----------------
extern "C" void kernel_launch(void* const* d_in, const int* in_sizes, int n_in,
                              void* d_out, int out_size, void* d_ws, size_t ws_size,
                              hipStream_t stream) {
  (void)in_sizes; (void)n_in; (void)out_size; (void)ws_size;
  const float* x_in = (const float*)d_in[0];
  const float* ec_s = (const float*)d_in[1];
  const float* ec_d = (const float*)d_in[2];
  const float* rc_e = (const float*)d_in[3];
  const float* rc_o = (const float*)d_in[4];
  const float* T0w  = (const float*)d_in[11];
  const float* T0b  = (const float*)d_in[12];

  char* ws = (char*)d_ws;
  float* XA = (float*)(ws + OFF_XA);
  float* XB = (float*)(ws + OFF_XB);
  float* Db = (float*)(ws + OFF_D);
  __bf16* xdr = (__bf16*)(ws + OFF_XFD_RE);
  __bf16* xdi = (__bf16*)(ws + OFF_XFD_IM);
  __bf16* xxr = (__bf16*)(ws + OFF_XFX_RE);
  __bf16* xxi = (__bf16*)(ws + OFF_XFX_IM);
  unsigned int* udft = (unsigned int*)(ws + OFF_UDFT);
  unsigned int* usft = (unsigned int*)(ws + OFF_USFT);
  float* UD = (float*)(ws + OFF_UD);
  float* US = (float*)(ws + OFF_US);
  __bf16* wb[6];
  for (int t = 0; t < 6; ++t) wb[t] = (__bf16*)(ws + OFF_WB + (size_t)t * WBSZ);

  // zero spectral-output buffers (inactive (m,level) tiles stay zero)
  {
    size_t cnt = (size_t)2 * NLEV * NBAT * CK * LMAX;
    k_fill0<<<dim3((unsigned)((cnt + 255) / 256)), dim3(256), 0, stream>>>(udft, cnt);
  }

  // convert the six spectral weight tensors to bf16 [m][o][i] (contiguous-K GEMM layout)
  for (int t = 0; t < 6; ++t)
    k_wconv<<<dim3(16, 2, 512), dim3(32, 8), 0, stream>>>((const float*)d_in[5 + t], wb[t]);

  // [b][n][i] -> [b][i][n]
  k_transpose_in<<<dim3(16, 128, 16), dim3(32, 8), 0, stream>>>(x_in, XA);

  // analysis chain + forward truncated DFT (per level)
  float* bufs[2] = { XA, XB };
  for (int j = 0; j < NLEV; ++j) {
    int halfN = (NSEQ >> j) >> 1;
    float* in  = bufs[j & 1];
    float* oxs = bufs[(j + 1) & 1];
    int total = NBAT * 64 * halfN;
    k_analysis<<<dim3((total + 255) / 256), dim3(256), 0, stream>>>(in, Db, oxs, ec_d, ec_s, halfN);
    int l = lmodes_hd(j);
    int mtiles = (l + 15) / 16;
    k_dft<<<dim3(mtiles, 32, 16), dim3(32), 0, stream>>>(Db,  xdr, xdi, halfN, j, l);
    k_dft<<<dim3(mtiles, 32, 16), dim3(32), 0, stream>>>(oxs, xxr, xxi, halfN, j, l);
  }

  // one weight pass for ALL levels (weights streamed once, bf16, contiguous)
  k_spectral<<<dim3(64, 32, 2), dim3(32), 0, stream>>>(xdr, xdi, xxr, xxi,
                                                       wb[0], wb[1], wb[2], wb[3], wb[4], wb[5],
                                                       udft, usft);

  // inverse truncated DFT -> Ud[j], Us[j] in reference [b][n][i] layout
  for (int j = 0; j < NLEV; ++j) {
    int Nn = NSEQ >> (j + 1);
    int l = lmodes_hd(j);
    int ntiles = Nn >= 16 ? Nn / 16 : 1;
    size_t off = (size_t)8192 * (size_t)(NSEQ - (NSEQ >> j));   // floats
    k_idft<<<dim3(ntiles, 32, 32), dim3(32), 0, stream>>>(udft, usft,
                                                          UD + off, US + off, Nn, j, l);
  }

  // coarsest-level linear; X12 sits in bufs[0] = XA
  float* R0 = Db;   // recon ping (analysis scratch is dead now)
  float* R1 = XB;   // recon pong
  k_t0<<<dim3((NBAT * CK + 255) / 256), dim3(256), 0, stream>>>(XA, T0w, T0b, R0);

  // synthesis chain
  float* cur = R0;
  for (int i = NLEV - 1; i >= 0; --i) {
    int len = NSEQ >> (i + 1);
    size_t off = (size_t)8192 * (size_t)(NSEQ - (NSEQ >> i));
    float* nxt = (i == 0) ? (float*)d_out : (cur == R0 ? R1 : R0);
    int total = NBAT * 64 * len;
    k_recon<<<dim3((total + 255) / 256), dim3(256), 0, stream>>>(cur, UD + off, US + off,
                                                                 rc_e, rc_o, nxt, len);
    cur = nxt;
  }
}
